// G_PDE_Solver2DLEMLinGated_88347477279254
// MI455X (gfx1250) — matrix operand order repaired and checked by reading, not compiled, not run on hardware
//
#include <hip/hip_runtime.h>

#define NN 8192
#define EE 131072
#define HH 128
#define TWD 25
#define NLAY 6

typedef _Float16 h16;
typedef __attribute__((ext_vector_type(8)))  _Float16 v8h;
typedef __attribute__((ext_vector_type(16))) _Float16 v16h;
typedef __attribute__((ext_vector_type(8)))  float    v8f;
typedef __attribute__((ext_vector_type(4)))  float    f4;

__device__ __forceinline__ float rcpf_(float x)     { return __builtin_amdgcn_rcpf(x); }
__device__ __forceinline__ float sigmoidf_(float x) { return rcpf_(1.0f + __expf(-x)); }
__device__ __forceinline__ float swishf_(float x)   { return x * sigmoidf_(x); }
// tanh(x) = 1 - 2/(e^{2x}+1); saturates correctly for |x| large
__device__ __forceinline__ float tanhf_(float x)    { return 1.0f - 2.0f * rcpf_(__expf(2.0f * x) + 1.0f); }
// order-preserving float->uint encoding for atomicMax
__device__ __forceinline__ unsigned encf_(float f) {
  int i = __float_as_int(f);
  return (i >= 0) ? ((unsigned)i ^ 0x80000000u) : ~(unsigned)i;
}
__device__ __forceinline__ float decf_(unsigned u) {
  int i = (u & 0x80000000u) ? (int)(u ^ 0x80000000u) : (int)~u;
  return __int_as_float(i);
}

// ---------------------------------------------------------------------------
// WMMA GEMM, all-f16 operands, f32 accumulate:
//   C[M,N] = act(A[M,K](f16) * W[N,K](f16)^T + bias[N]),  optional f16 mirror C16.
// Fragments loaded straight from global as contiguous 16B runs (documented
// wave32 f16 A 16x32 / B 32x16 VGPR layouts are contiguous per lane group).
// Block = 256 threads = 8 waves; each wave owns one 16x16 C tile of a
// 16(M) x 128(N) block tile. No LDS, no barriers.
// act: 0 = none, 1 = swish
// ---------------------------------------------------------------------------
__global__ __launch_bounds__(256) void k_gemm(
    const h16* __restrict__ A, const h16* __restrict__ W,
    const float* __restrict__ bias, float* __restrict__ C,
    h16* __restrict__ C16, int M, int N, int K, int act)
{
  const int tid  = threadIdx.x;
  const int lane = tid & 31;
  const int wave = tid >> 5;
  const int lg   = lane >> 4;     // lane group (0/1)
  const int l15  = lane & 15;
  const int m0 = blockIdx.x * 16;
  const int n0 = (blockIdx.y * 8 + wave) * 16;
  if (n0 >= N || m0 >= M) return;

  const h16* __restrict__ arow = A + (size_t)(m0 + l15) * K;  // A row for this lane
  const h16* __restrict__ brow = W + (size_t)(n0 + l15) * K;  // W row (= B column)

  v8f acc = {};
#pragma unroll
  for (int k0 = 0; k0 < 128; k0 += 32) {   // K == 128 for every GEMM in this net
    // A 16x32 f16 frag: halves 0..7 <- K[k0+8g .. +7], 8..15 <- K[k0+16+8g .. +7]
    v8h a_lo = *(const v8h*)(arow + k0 + 8 * lg);
    v8h a_hi = *(const v8h*)(arow + k0 + 16 + 8 * lg);
    // B 32x16 f16 frag: halves 0..7 <- K[k0+8g .. +7], 8..15 <- K[k0+16+8g .. +7]
    v8h b_lo = *(const v8h*)(brow + k0 + 8 * lg);
    v8h b_hi = *(const v8h*)(brow + k0 + 16 + 8 * lg);
    v16h af = __builtin_shufflevector(a_lo, a_hi, 0,1,2,3,4,5,6,7,8,9,10,11,12,13,14,15);
    v16h bf = __builtin_shufflevector(b_lo, b_hi, 0,1,2,3,4,5,6,7,8,9,10,11,12,13,14,15);
    acc = __builtin_amdgcn_wmma_f32_16x16x32_f16(false, af, false, bf,
                                                 (short)0, acc, false, false);
  }

  // C layout: VGPR i -> M = i + 8*laneGroup ; N = lane&15
  const int n = n0 + l15;
  const float bv = bias ? bias[n] : 0.0f;
#pragma unroll
  for (int i = 0; i < 8; ++i) {
    int m = m0 + i + lg * 8;
    float v = acc[i] + bv;
    if (act == 1) v = swishf_(v);
    C[(size_t)m * N + n] = v;
    if (C16) C16[(size_t)m * N + n] = (h16)v;
  }
}

// ---------------------------------------------------------------------------
// Utility kernels
// ---------------------------------------------------------------------------
__global__ void k_f32_to_f16(const float* __restrict__ s, h16* __restrict__ d, int n) {
  int i = blockIdx.x * blockDim.x + threadIdx.x;
  if (i < n) d[i] = (h16)s[i];
}
__global__ void k_fill_u32(unsigned* __restrict__ p, unsigned v, int n) {
  int i = blockIdx.x * blockDim.x + threadIdx.x;
  if (i < n) p[i] = v;
}

// edge_attr[e, 0..50] = concat(u[src]-u[dst] (50), pos_x diff (1))
__global__ void k_edge_attr(const int* __restrict__ ei, const float* __restrict__ u,
                            const float* __restrict__ pos, float* __restrict__ attr) {
  int idx = blockIdx.x * blockDim.x + threadIdx.x;
  if (idx >= EE * 51) return;
  int e = idx / 51, c = idx - e * 51;
  int s = ei[e], d = ei[EE + e];
  float v;
  if (c < 50) v = u[(size_t)s * 50 + c] - u[(size_t)d * 50 + c];
  else        v = (pos[2 * s + 1] - pos[2 * d + 1]) * (1.0f / 16.0f);
  attr[idx] = v;
}

// LEM elementwise part A: gates + z update (also writes f16 mirror of z).
__global__ void k_lem_A(int t, const float* __restrict__ u, const float* __restrict__ pos,
                        const float* __restrict__ iw, const float* __restrict__ ib,
                        const float* __restrict__ hb, const float* __restrict__ zb,
                        const float* __restrict__ hh, float* __restrict__ z,
                        h16* __restrict__ z16,
                        float* __restrict__ msbar, float* __restrict__ i4b) {
  int idx = blockIdx.x * blockDim.x + threadIdx.x;
  if (idx >= NN * HH) return;
  int n = idx >> 7, j = idx & 127;
  float x0 = pos[2 * n + 1] * (1.0f / 16.0f);
  float x1 = u[(size_t)n * 50 + t];
  float x2 = u[(size_t)n * 50 + 25 + t];
  float x3 = pos[2 * n] * 0.25f + 0.04f * (float)(t + 1);
  float ii[4];
#pragma unroll
  for (int q = 0; q < 4; ++q) {
    int r = q * HH + j;
    f4 wr = *(const f4*)(iw + r * 4);
    ii[q] = wr.x * x0 + wr.y * x1 + wr.z * x2 + wr.w * x3 + ib[r];
  }
  float h1 = hh[(size_t)n * 384 + j]        + hb[j];
  float h2 = hh[(size_t)n * 384 + 128 + j]  + hb[128 + j];
  float h3 = hh[(size_t)n * 384 + 256 + j]  + hb[256 + j];
  float msb = sigmoidf_(ii[0] + h1);
  float msd = sigmoidf_(ii[1] + h2);
  float zn = (1.0f - msd) * z[idx] + msd * tanhf_(ii[2] + h3);
  z[idx]   = zn;
  z16[idx] = (h16)zn;
  msbar[idx] = msb;
  i4b[idx]   = ii[3] + zb[j];   // fold in lem_z_b
}

// LEM elementwise part B: y update using g = z@W_z^T (also writes f16 mirror of y).
__global__ void k_lem_B(const float* __restrict__ g, const float* __restrict__ msbar,
                        const float* __restrict__ i4b, float* __restrict__ y,
                        h16* __restrict__ y16) {
  int idx = blockIdx.x * blockDim.x + threadIdx.x;
  if (idx >= NN * HH) return;
  float msb = msbar[idx];
  float yn = (1.0f - msb) * y[idx] + msb * tanhf_(g[idx] + i4b[idx]);
  y[idx]   = yn;
  y16[idx] = (h16)yn;
}

// per-node attention scalars: xt . a_src, xt . a_dst for gate and main
__global__ void k_att_scalars(const float* __restrict__ xtg, const float* __restrict__ xtm,
                              const float* __restrict__ gas, const float* __restrict__ gad,
                              const float* __restrict__ mas, const float* __restrict__ mad,
                              float* __restrict__ asg, float* __restrict__ adg,
                              float* __restrict__ asm2, float* __restrict__ adm) {
  int n = blockIdx.x * blockDim.x + threadIdx.x;
  if (n >= NN) return;
  float sg = 0.f, dg = 0.f, sm = 0.f, dm = 0.f;
  const f4* xg = (const f4*)(xtg + (size_t)n * HH);
  const f4* xm = (const f4*)(xtm + (size_t)n * HH);
  const f4* g1 = (const f4*)gas;
  const f4* g2 = (const f4*)gad;
  const f4* m1 = (const f4*)mas;
  const f4* m2 = (const f4*)mad;
  for (int j = 0; j < HH / 4; ++j) {
    f4 a = xg[j], b = xm[j], p = g1[j], q = g2[j], r = m1[j], s = m2[j];
    sg += a.x*p.x + a.y*p.y + a.z*p.z + a.w*p.w;
    dg += a.x*q.x + a.y*q.y + a.z*q.z + a.w*q.w;
    sm += b.x*r.x + b.y*r.y + b.z*r.z + b.w*r.w;
    dm += b.x*s.x + b.y*s.y + b.z*s.z + b.w*s.w;
  }
  asg[n] = sg; adg[n] = dg; asm2[n] = sm; adm[n] = dm;
}

// v51 = edge_w.T @ a_edge   (et . a_edge == edge_attr . v51)
__global__ void k_v51(const float* __restrict__ gae, const float* __restrict__ gew,
                      const float* __restrict__ mae, const float* __restrict__ mew,
                      float* __restrict__ vg, float* __restrict__ vm) {
  int d = threadIdx.x;
  if (d >= 51) return;
  float a = 0.f, b = 0.f;
  for (int h = 0; h < HH; ++h) {
    a += gae[h] * gew[h * 51 + d];
    b += mae[h] * mew[h * 51 + d];
  }
  vg[d] = a; vm[d] = b;
}

// pass 1: logits + atomic max per dst
__global__ void k_edge1(const int* __restrict__ ei, const float* __restrict__ attr,
                        const float* __restrict__ asg, const float* __restrict__ adg,
                        const float* __restrict__ asm2, const float* __restrict__ adm,
                        const float* __restrict__ vg, const float* __restrict__ vm,
                        float* __restrict__ logg, float* __restrict__ logm,
                        unsigned* __restrict__ amg, unsigned* __restrict__ amm) {
  int e = blockIdx.x * blockDim.x + threadIdx.x;
  if (e >= EE) return;
  int s = ei[e], d = ei[EE + e];
  const float* ar = attr + (size_t)e * 51;
  float dg = 0.f, dm = 0.f;
  for (int c = 0; c < 51; ++c) { float a = ar[c]; dg += a * vg[c]; dm += a * vm[c]; }
  float lg = asg[s] + adg[d] + dg; lg = lg > 0.f ? lg : 0.2f * lg;
  float lm = asm2[s] + adm[d] + dm; lm = lm > 0.f ? lm : 0.2f * lm;
  logg[e] = lg; logm[e] = lm;
  atomicMax(amg + d, encf_(lg));
  atomicMax(amm + d, encf_(lm));
}

// pass 2: exp(logit - max) + atomic sum per dst
__global__ void k_edge2(const int* __restrict__ ei,
                        const float* __restrict__ logg, const float* __restrict__ logm,
                        const unsigned* __restrict__ amg, const unsigned* __restrict__ amm,
                        float* __restrict__ exg, float* __restrict__ exm,
                        float* __restrict__ deng, float* __restrict__ denm) {
  int e = blockIdx.x * blockDim.x + threadIdx.x;
  if (e >= EE) return;
  int d = ei[EE + e];
  float eg = __expf(logg[e] - decf_(amg[d]));
  float em = __expf(logm[e] - decf_(amm[d]));
  exg[e] = eg; exm[e] = em;
  atomicAdd(deng + d, eg);
  atomicAdd(denm + d, em);
}

// pass 3: weighted gather/scatter of xt[src] -> agg[dst]; 128 channels per edge
__global__ __launch_bounds__(256) void k_edge3(
    const int* __restrict__ ei,
    const float* __restrict__ exg, const float* __restrict__ exm,
    const float* __restrict__ deng, const float* __restrict__ denm,
    const float* __restrict__ xtg, const float* __restrict__ xtm,
    float* __restrict__ aggg, float* __restrict__ aggm) {
  int t = threadIdx.x;
  int e = blockIdx.x * 2 + (t >> 7);
  int c = t & 127;
  if (e >= EE) return;
  int s = ei[e], d = ei[EE + e];
  float wg = exg[e] * rcpf_(deng[d] + 1e-16f);
  float wm = exm[e] * rcpf_(denm[d] + 1e-16f);
  atomicAdd(aggg + (size_t)d * HH + c, wg * xtg[(size_t)s * HH + c]);
  atomicAdd(aggm + (size_t)d * HH + c, wm * xtm[(size_t)s * HH + c]);
}

// gated combine: h = (1-tau)*h + tau*swish(msg)  (also refresh f16 mirror of h)
__global__ void k_combine(float* __restrict__ h, h16* __restrict__ hf,
                          const float* __restrict__ aggg, const float* __restrict__ aggm,
                          const float* __restrict__ gb, const float* __restrict__ mb) {
  int idx = blockIdx.x * blockDim.x + threadIdx.x;
  if (idx >= NN * HH) return;
  int c = idx & 127;
  float tau = sigmoidf_(aggg[idx] + gb[c]);
  float msg = swishf_(aggm[idx] + mb[c]);
  float hn = (1.0f - tau) * h[idx] + tau * msg;
  h[idx]  = hn;
  hf[idx] = (h16)hn;
}

// decoder: per-node conv1d (2->8, k16, s3) + swish + conv1d (8->2, k14) ; out = u + dt*d
__global__ __launch_bounds__(128) void k_decoder(
    const float* __restrict__ h2, const float* __restrict__ u,
    const float* __restrict__ c1w, const float* __restrict__ c1b,
    const float* __restrict__ c2w, const float* __restrict__ c2b,
    float* __restrict__ out) {
  __shared__ float sh2[256];
  __shared__ float sd1[8][38];
  int n = blockIdx.x, t = threadIdx.x;
  for (int i = t; i < 256; i += 128) sh2[i] = h2[(size_t)n * 256 + i];
  __syncthreads();
  for (int idx = t; idx < 8 * 38; idx += 128) {
    int o = idx / 38, p = idx - o * 38;
    float s = c1b[o];
    for (int ic = 0; ic < 2; ++ic)
      for (int k = 0; k < 16; ++k)
        s += sh2[ic * 128 + 3 * p + k] * c1w[(o * 2 + ic) * 16 + k];
    sd1[o][p] = swishf_(s);
  }
  __syncthreads();
  for (int idx = t; idx < 50; idx += 128) {
    int o = idx / 25, p = idx - o * 25;
    float s = c2b[o];
    for (int ic = 0; ic < 8; ++ic)
      for (int k = 0; k < 14; ++k)
        s += sd1[ic][p + k] * c2w[(o * 8 + ic) * 14 + k];
    size_t oi = (size_t)n * 50 + o * 25 + p;
    out[oi] = u[oi] + 0.04f * (float)(p + 1) * s;
  }
}

// ---------------------------------------------------------------------------
extern "C" void kernel_launch(void* const* d_in, const int* in_sizes, int n_in,
                              void* d_out, int out_size, void* d_ws, size_t ws_size,
                              hipStream_t stream) {
  const float* u          = (const float*)d_in[0];
  const float* pos        = (const float*)d_in[1];
  const int*   ei         = (const int*)  d_in[2];
  const float* lem_inp_w  = (const float*)d_in[3];
  const float* lem_inp_b  = (const float*)d_in[4];
  const float* lem_hid_w  = (const float*)d_in[5];
  const float* lem_hid_b  = (const float*)d_in[6];
  const float* lem_z_w    = (const float*)d_in[7];
  const float* lem_z_b    = (const float*)d_in[8];
  const float* mlp1_w     = (const float*)d_in[9];
  const float* mlp1_b     = (const float*)d_in[10];
  const float* mlp2_w     = (const float*)d_in[11];
  const float* mlp2_b     = (const float*)d_in[12];
  const float* gnn_lin_w  = (const float*)d_in[13];
  const float* gnn_edge_w = (const float*)d_in[14];
  const float* gnn_att_src  = (const float*)d_in[15];
  const float* gnn_att_dst  = (const float*)d_in[16];
  const float* gnn_att_edge = (const float*)d_in[17];
  const float* gnn_bias   = (const float*)d_in[18];
  const float* gate_lin_w = (const float*)d_in[19];
  const float* gate_edge_w= (const float*)d_in[20];
  const float* gate_att_src  = (const float*)d_in[21];
  const float* gate_att_dst  = (const float*)d_in[22];
  const float* gate_att_edge = (const float*)d_in[23];
  const float* gate_bias  = (const float*)d_in[24];
  const float* dbl_w = (const float*)d_in[25];
  const float* dbl_b = (const float*)d_in[26];
  const float* c1w = (const float*)d_in[27];
  const float* c1b = (const float*)d_in[28];
  const float* c2w = (const float*)d_in[29];
  const float* c2b = (const float*)d_in[30];
  float* out = (float*)d_out;

  // ---- workspace bump allocator ----
  char* base = (char*)d_ws;
  size_t off = 0;
  auto alloc = [&](size_t bytes) -> void* {
    void* p = base + off;
    off += (bytes + 255) & ~(size_t)255;
    return p;
  };

  // f16 weight copies ([out,in] row-major, i.e. [N,K])
  h16* w_hid  = (h16*)alloc(384 * 128 * 2);
  h16* w_z    = (h16*)alloc(128 * 128 * 2);
  h16* w_m1   = (h16*)alloc(128 * 128 * 2);
  h16* w_m2   = (h16*)alloc(128 * 128 * 2);
  h16* w_gnn  = (h16*)alloc(NLAY * 128 * 128 * 2);
  h16* w_gate = (h16*)alloc(NLAY * 128 * 128 * 2);
  h16* w_dbl  = (h16*)alloc(256 * 128 * 2);

  const int NH = NN * HH;
  float* attr  = (float*)alloc((size_t)EE * 51 * 4);
  float* yz    = (float*)alloc((size_t)2 * NH * 4);   // y, z contiguous (one zero-fill)
  float* y     = yz;
  float* z     = yz + (size_t)NH;
  h16*  yz16   = (h16*)alloc((size_t)2 * NH * 2);     // f16 mirrors of y, z
  h16*  y16    = yz16;
  h16*  z16    = yz16 + (size_t)NH;
  h16*  tmp16  = (h16*)alloc((size_t)NH * 2);
  h16*  h16buf = (h16*)alloc((size_t)NH * 2);
  float* msbar = (float*)alloc((size_t)NH * 4);
  float* i4b   = (float*)alloc((size_t)NH * 4);
  float* hh    = (float*)alloc((size_t)NN * 384 * 4);
  float* gbuf  = (float*)alloc((size_t)NH * 4);
  float* tmp   = (float*)alloc((size_t)NH * 4);
  float* h     = (float*)alloc((size_t)NH * 4);
  float* xtg   = (float*)alloc((size_t)NH * 4);
  float* xtm   = (float*)alloc((size_t)NH * 4);
  float* agg2  = (float*)alloc((size_t)2 * NH * 4);
  float* aggg  = agg2;
  float* aggm  = agg2 + (size_t)NH;
  float* h2buf = (float*)alloc((size_t)NN * 256 * 4);
  float* logg  = (float*)alloc((size_t)EE * 4);
  float* logm  = (float*)alloc((size_t)EE * 4);
  float* exg   = (float*)alloc((size_t)EE * 4);
  float* exm   = (float*)alloc((size_t)EE * 4);
  float* asg   = (float*)alloc(NN * 4);
  float* adg   = (float*)alloc(NN * 4);
  float* asm2  = (float*)alloc(NN * 4);
  float* adm   = (float*)alloc(NN * 4);
  unsigned* am2 = (unsigned*)alloc((size_t)2 * NN * 4);
  unsigned* amg = am2;
  unsigned* amm = am2 + NN;
  float* den2 = (float*)alloc((size_t)2 * NN * 4);
  float* deng = den2;
  float* denm = den2 + NN;
  float* v51g = (float*)alloc(64 * 4);
  float* v51m = (float*)alloc(64 * 4);
  (void)ws_size; (void)in_sizes; (void)n_in; (void)out_size;

  auto cvt = [&](const float* s, h16* d, int n) {
    k_f32_to_f16<<<(n + 255) / 256, 256, 0, stream>>>(s, d, n);
  };
  auto fill = [&](void* p, unsigned v, int n) {
    k_fill_u32<<<(n + 255) / 256, 256, 0, stream>>>((unsigned*)p, v, n);
  };
  auto gemm = [&](const h16* A, const h16* W, const float* bias, float* C, h16* C16,
                  int M, int Nc, int K, int act) {
    k_gemm<<<dim3(M / 16, (Nc + 127) / 128), 256, 0, stream>>>(A, W, bias, C, C16,
                                                               M, Nc, K, act);
  };

  // ---- weight conversion ----
  cvt(lem_hid_w, w_hid, 384 * 128);
  cvt(lem_z_w,   w_z,   128 * 128);
  cvt(mlp1_w,    w_m1,  128 * 128);
  cvt(mlp2_w,    w_m2,  128 * 128);
  cvt(gnn_lin_w,  w_gnn,  NLAY * 128 * 128);
  cvt(gate_lin_w, w_gate, NLAY * 128 * 128);
  cvt(dbl_w,      w_dbl,  256 * 128);

  // ---- edge attributes (shared by all layers) ----
  k_edge_attr<<<(EE * 51 + 255) / 256, 256, 0, stream>>>(ei, u, pos, attr);

  // ---- LEM scan ----
  fill(yz,   0u, 2 * NH);       // f32 y,z = 0
  fill(yz16, 0u, NH);           // f16 y,z = 0 (2*NH halves == NH dwords)
  for (int t = 0; t < TWD; ++t) {
    gemm(y16, w_hid, nullptr, hh, nullptr, NN, 384, 128, 0);
    k_lem_A<<<(NH + 255) / 256, 256, 0, stream>>>(t, u, pos, lem_inp_w, lem_inp_b,
                                                  lem_hid_b, lem_z_b, hh, z, z16,
                                                  msbar, i4b);
    gemm(z16, w_z, nullptr, gbuf, nullptr, NN, 128, 128, 0);
    k_lem_B<<<(NH + 255) / 256, 256, 0, stream>>>(gbuf, msbar, i4b, y, y16);
  }

  // ---- output MLP ----
  gemm(y16,   w_m1, mlp1_b, tmp, tmp16,  NN, 128, 128, 1);
  gemm(tmp16, w_m2, mlp2_b, h,   h16buf, NN, 128, 128, 1);

  // ---- gated GAT layers ----
  for (int i = 0; i < NLAY; ++i) {
    gemm(h16buf, w_gate + (size_t)i * 128 * 128, nullptr, xtg, nullptr, NN, 128, 128, 0);
    gemm(h16buf, w_gnn  + (size_t)i * 128 * 128, nullptr, xtm, nullptr, NN, 128, 128, 0);
    k_att_scalars<<<(NN + 255) / 256, 256, 0, stream>>>(
        xtg, xtm, gate_att_src + i * HH, gate_att_dst + i * HH,
        gnn_att_src + i * HH, gnn_att_dst + i * HH, asg, adg, asm2, adm);
    k_v51<<<1, 64, 0, stream>>>(gate_att_edge + i * HH, gate_edge_w + (size_t)i * HH * 51,
                                gnn_att_edge + i * HH,  gnn_edge_w + (size_t)i * HH * 51,
                                v51g, v51m);
    fill(am2,  0u, 2 * NN);            // encoded -inf sentinel (unsigned 0 is minimum)
    fill(den2, 0u, 2 * NN);
    fill(agg2, 0u, 2 * NH);
    k_edge1<<<(EE + 255) / 256, 256, 0, stream>>>(ei, attr, asg, adg, asm2, adm,
                                                  v51g, v51m, logg, logm, amg, amm);
    k_edge2<<<(EE + 255) / 256, 256, 0, stream>>>(ei, logg, logm, amg, amm,
                                                  exg, exm, deng, denm);
    k_edge3<<<EE / 2, 256, 0, stream>>>(ei, exg, exm, deng, denm, xtg, xtm, aggg, aggm);
    k_combine<<<(NH + 255) / 256, 256, 0, stream>>>(h, h16buf, aggg, aggm,
                                                    gate_bias + i * HH, gnn_bias + i * HH);
  }

  // ---- decoder ----
  gemm(h16buf, w_dbl, dbl_b, h2buf, nullptr, NN, 256, 128, 1);
  k_decoder<<<NN, 128, 0, stream>>>(h2buf, u, c1w, c1b, c2w, c2b, out);
}
